// NaiveLSTM_45011257262622
// MI455X (gfx1250) — matrix operand order repaired and checked by reading, not compiled, not run on hardware
//
#include <hip/hip_runtime.h>
#include <hip/hip_bf16.h>

typedef __attribute__((ext_vector_type(16))) _Float16 v16h;
typedef __attribute__((ext_vector_type(8)))  _Float16 v8h;
typedef __attribute__((ext_vector_type(8)))  float    v8f;

#define B_  256
#define T_  2048
#define I_  128
#define H_  128
// M = B*T = 524288 rows, K = I = 128, N = H = 128, 4 gates (i,f,g,o)

__device__ __forceinline__ float fsig(float x) {
    return 1.0f / (1.0f + __expf(-x));
}
__device__ __forceinline__ float ftanh(float x) {
    float ax = fabsf(x);
    float e  = __expf(-2.0f * ax);
    float t  = (1.0f - e) / (1.0f + e);
    return copysignf(t, x);
}

// ---------------------------------------------------------------------------
// Prep 1: gbias[g][n] = b_i[n] + b_h[n] + sum_k h[k] * w_h[k*H+n]
// One block of 512 threads (gate = tid>>7, n = tid&127).
// ---------------------------------------------------------------------------
__global__ void lstm_prep_bias(
    const float* __restrict__ h,
    const float* __restrict__ whi, const float* __restrict__ bii, const float* __restrict__ bhi,
    const float* __restrict__ whf, const float* __restrict__ bif, const float* __restrict__ bhf,
    const float* __restrict__ whg, const float* __restrict__ big, const float* __restrict__ bhg,
    const float* __restrict__ who, const float* __restrict__ bio, const float* __restrict__ bho,
    float* __restrict__ gbias)
{
    int tid = threadIdx.x;               // 0..511
    int g = tid >> 7, n = tid & (H_ - 1);
    const float* wh = (g == 0) ? whi : (g == 1) ? whf : (g == 2) ? whg : who;
    const float* bi = (g == 0) ? bii : (g == 1) ? bif : (g == 2) ? big : bio;
    const float* bh = (g == 0) ? bhi : (g == 1) ? bhf : (g == 2) ? bhg : bho;
    float s = bi[n] + bh[n];
    #pragma unroll 8
    for (int k = 0; k < H_; ++k) s += h[k] * wh[k * H_ + n];
    gbias[g * H_ + n] = s;
}

// ---------------------------------------------------------------------------
// Prep 2: pack the 4 input weight matrices (I x H, f32, row-major = K-major)
// into f16 WMMA B-fragments:
//   packB[nt][g][kt][lane][e], e=0..15:  value = w_g[K*H + n]
//   where K = kt*32 + (lane>>4)*16 + e,  n = nt*16 + (lane&15)
// Total 4 gates * 8 ntiles * 4 ktiles * 32 lanes * 16 = 65536 f16 = 128 KB.
// ---------------------------------------------------------------------------
__global__ void lstm_pack_b(
    const float* __restrict__ wii, const float* __restrict__ wif,
    const float* __restrict__ wig, const float* __restrict__ wio,
    _Float16* __restrict__ packB)
{
    int idx  = blockIdx.x * 256 + threadIdx.x;   // 0..65535
    int e    = idx & 15;
    int lane = (idx >> 4) & 31;
    int kt   = (idx >> 9) & 3;
    int g    = (idx >> 11) & 3;
    int nt   = idx >> 13;
    const float* w = (g == 0) ? wii : (g == 1) ? wif : (g == 2) ? wig : wio;
    int K = kt * 32 + (lane >> 4) * 16 + e;
    int n = nt * 16 + (lane & 15);
    packB[idx] = (_Float16)w[K * H_ + n];
}

// ---------------------------------------------------------------------------
// Main: one workgroup (8 waves, 256 threads) per 16-row M tile.
//  - Stage the 16x128 f32 A tile into LDS as f16 in A-fragment order (4 KB).
//  - Wave w owns N-tile w (16 cols): 4 accumulators (gates i,f,g,o),
//    4 k-steps of v_wmma_f32_16x16x32_f16 each.
//  - Fuse bias + sigmoid/tanh + LSTM cell math + transposed store.
// ---------------------------------------------------------------------------
__global__ __launch_bounds__(256) void lstm_main(
    const float*    __restrict__ inputs,
    const _Float16* __restrict__ packB,
    const float*    __restrict__ gbias,
    const float*    __restrict__ cvec,
    float*          __restrict__ out)
{
    __shared__ __attribute__((aligned(32))) _Float16 sA[4 * 32 * 16]; // 4 KB

    const int tid   = threadIdx.x;
    const int mtile = blockIdx.x;               // 0..32767

    // ---- Stage A tile: thread handles 8 consecutive fragment f16 elems ----
    {
        int fidx = tid * 8;                      // flat fragment index
        int e0   = (tid & 1) * 8;                // 0 or 8
        int lane = (tid >> 1) & 31;
        int kt   = tid >> 6;
        int row  = lane & 15;
        // A-fragment K mapping: K(e) = kb + e, kb = 16*(lane>>4)
        int Kst  = kt * 32 + (lane >> 4) * 16 + (e0 ? 8 : 0);
        const float* src = inputs + (size_t)(mtile * 16 + row) * I_ + Kst;
        float4 f0 = *(const float4*)(src);
        float4 f1 = *(const float4*)(src + 4);
        v8h hv;
        hv[0] = (_Float16)f0.x; hv[1] = (_Float16)f0.y;
        hv[2] = (_Float16)f0.z; hv[3] = (_Float16)f0.w;
        hv[4] = (_Float16)f1.x; hv[5] = (_Float16)f1.y;
        hv[6] = (_Float16)f1.z; hv[7] = (_Float16)f1.w;
        *(v8h*)(sA + fidx) = hv;                 // 16B-aligned ds_store
    }
    __syncthreads();

    // ---- Per-wave GEMM: wave id = N-tile ----
    const int lane = tid & 31;
    const int nt   = tid >> 5;                   // 0..7
    const _Float16* bB = packB + (size_t)nt * 8192; // [g][kt][lane][16]

    v8f accI = {}, accF = {}, accG = {}, accO = {};
    #pragma unroll
    for (int kt = 0; kt < 4; ++kt) {
        v16h a  = *(const v16h*)(sA + (kt * 32 + lane) * 16);
        v16h b0 = *(const v16h*)(bB + ((0 * 4 + kt) * 32 + lane) * 16);
        v16h b1 = *(const v16h*)(bB + ((1 * 4 + kt) * 32 + lane) * 16);
        v16h b2 = *(const v16h*)(bB + ((2 * 4 + kt) * 32 + lane) * 16);
        v16h b3 = *(const v16h*)(bB + ((3 * 4 + kt) * 32 + lane) * 16);
        accI = __builtin_amdgcn_wmma_f32_16x16x32_f16(false, a, false, b0, (short)0, accI, false, false);
        accF = __builtin_amdgcn_wmma_f32_16x16x32_f16(false, a, false, b1, (short)0, accF, false, false);
        accG = __builtin_amdgcn_wmma_f32_16x16x32_f16(false, a, false, b2, (short)0, accG, false, false);
        accO = __builtin_amdgcn_wmma_f32_16x16x32_f16(false, a, false, b3, (short)0, accO, false, false);
    }

    // ---- Fused LSTM cell math + transposed stores ----
    const int n  = nt * 16 + (lane & 15);        // C/D: N = lane%16
    const float bi = gbias[0 * H_ + n];
    const float bf = gbias[1 * H_ + n];
    const float bg = gbias[2 * H_ + n];
    const float bo = gbias[3 * H_ + n];
    const float c0 = cvec[n];
    const int mbase = mtile * 16 + (lane >> 4) * 8;  // C/D: M = r + 8*(lane>>4)
    const size_t TBH = (size_t)T_ * B_ * H_;

    #pragma unroll
    for (int r = 0; r < 8; ++r) {
        int m = mbase + r;
        int b = m >> 11;            // m / T_
        int t = m & (T_ - 1);       // m % T_
        float xi = accI[r] + bi;
        float xf = accF[r] + bf;
        float xg = accG[r] + bg;
        float xo = accO[r] + bo;
        float ig = fsig(xi), fg = fsig(xf), og = fsig(xo), gg = ftanh(xg);
        float cn = fg * c0 + ig * gg;
        float hn = og * ftanh(cn);
        out[((size_t)t * B_ + b) * H_ + n] = hn;       // hidden_seq (T,B,1,H)
        if (t == T_ - 1) {
            out[TBH + (size_t)b * H_ + n]                    = hn;  // h_last
            out[TBH + (size_t)B_ * H_ + (size_t)b * H_ + n]  = cn;  // c_last
        }
    }
}

// ---------------------------------------------------------------------------
extern "C" void kernel_launch(void* const* d_in, const int* in_sizes, int n_in,
                              void* d_out, int out_size, void* d_ws, size_t ws_size,
                              hipStream_t stream) {
    (void)in_sizes; (void)n_in; (void)out_size; (void)ws_size;
    const float* inputs = (const float*)d_in[0];
    const float* h      = (const float*)d_in[1];
    const float* c      = (const float*)d_in[2];
    const float* w_ii = (const float*)d_in[3];  const float* w_hi = (const float*)d_in[4];
    const float* b_ii = (const float*)d_in[5];  const float* b_hi = (const float*)d_in[6];
    const float* w_if = (const float*)d_in[7];  const float* w_hf = (const float*)d_in[8];
    const float* b_if = (const float*)d_in[9];  const float* b_hf = (const float*)d_in[10];
    const float* w_io = (const float*)d_in[11]; const float* w_ho = (const float*)d_in[12];
    const float* b_io = (const float*)d_in[13]; const float* b_ho = (const float*)d_in[14];
    const float* w_ig = (const float*)d_in[15]; const float* w_hg = (const float*)d_in[16];
    const float* b_ig = (const float*)d_in[17]; const float* b_hg = (const float*)d_in[18];

    _Float16* packB = (_Float16*)d_ws;                   // 128 KB
    float*    gbias = (float*)((char*)d_ws + 131072);    // 2 KB
    float*    out   = (float*)d_out;

    lstm_prep_bias<<<1, 512, 0, stream>>>(h,
        w_hi, b_ii, b_hi,
        w_hf, b_if, b_hf,
        w_hg, b_ig, b_hg,
        w_ho, b_io, b_ho,
        gbias);
    lstm_pack_b<<<256, 256, 0, stream>>>(w_ii, w_if, w_ig, w_io, packB);

    const int num_mtiles = (B_ * T_) / 16;               // 32768
    lstm_main<<<num_mtiles, 256, 0, stream>>>(inputs, packB, gbias, c, out);
}
